// MoDBlock_18021682774285
// MI455X (gfx1250) — compile-verified, matrix operand order checked by reading, not run on hardware
//
#include <hip/hip_runtime.h>
#include <hip/hip_bf16.h>
#include <math.h>

// ---------------- problem constants ----------------
#define B_   2
#define S_   2048
#define H_   1024
#define F_   4096
#define NH_  16
#define HD_  64
#define CAP_ 1024                   // int(S * 0.5)
#define BS_  (B_ * S_)              // 4096
#define BSH_ ((size_t)BS_ * H_)     // 4,194,304
#define BSF_ ((size_t)BS_ * F_)     // 16,777,216
#define EPS_ 1e-6f

typedef __attribute__((ext_vector_type(16))) __bf16 v16bf;
typedef __attribute__((ext_vector_type(8)))  float  v8f;

union FragBF { v16bf v; uint4 q[2]; };

__device__ __forceinline__ v8f wmma_bf16f32(v16bf a, v16bf b, v8f c) {
  // D(f32 16x16) = A(bf16 16x32) x B(bf16 32x16) + C
  return __builtin_amdgcn_wmma_f32_16x16x32_bf16(false, a, false, b, (short)0, c, false, false);
}

__device__ __forceinline__ float halfmax16(float v) {
  v = fmaxf(v, __shfl_xor(v, 1, 32));
  v = fmaxf(v, __shfl_xor(v, 2, 32));
  v = fmaxf(v, __shfl_xor(v, 4, 32));
  v = fmaxf(v, __shfl_xor(v, 8, 32));
  return v;
}
__device__ __forceinline__ float halfsum16(float v) {
  v += __shfl_xor(v, 1, 32);
  v += __shfl_xor(v, 2, 32);
  v += __shfl_xor(v, 4, 32);
  v += __shfl_xor(v, 8, 32);
  return v;
}

// -------- gfx1250 async global->LDS copy (guarded; 16B per lane) --------
#if defined(__has_builtin)
#if __has_builtin(__builtin_amdgcn_global_load_async_to_lds_b128)
#define ASYNC_LDS 1
#endif
#endif
#ifndef ASYNC_LDS
#define ASYNC_LDS 0
#endif

// pointer types matching the builtin's declared params:
//   param0: v4i in AS1 (global), param1: v4i in AS3 (LDS)
typedef __attribute__((vector_size(16))) int vsi4;
typedef __attribute__((address_space(1))) vsi4* gvec_p;
typedef __attribute__((address_space(3))) vsi4* lvec_p;

__device__ __forceinline__ void stage16(const __bf16* g, __bf16* l) {
#if ASYNC_LDS
  __builtin_amdgcn_global_load_async_to_lds_b128((gvec_p)g, (lvec_p)l, 0, 0);
#else
  *(uint4*)l = *(const uint4*)g;  // global_load_b128 + ds_store_b128 fallback
#endif
}
__device__ __forceinline__ void stage_wait() {
#if ASYNC_LDS
  asm volatile("s_wait_asynccnt 0" ::: "memory");
#else
  asm volatile("s_wait_dscnt 0" ::: "memory");
#endif
}

// ---------------- elementwise helpers ----------------
__global__ void cast_bf16_k(const float* __restrict__ in, __bf16* __restrict__ out, size_t n) {
  size_t i = (size_t)blockIdx.x * blockDim.x + threadIdx.x;
  if (i < n) out[i] = (__bf16)in[i];
}

// O[i] = R[i] + mask[row] * C[i]   (row = i / H_)
__global__ void resadd_k(const float* __restrict__ R, const float* __restrict__ C,
                         const float* __restrict__ Mk, float* __restrict__ O, size_t n) {
  size_t i = (size_t)blockIdx.x * blockDim.x + threadIdx.x;
  if (i >= n) return;
  O[i] = R[i] + Mk[i >> 10] * C[i];
}

// ---------------- router + top-k mask ----------------
__global__ __launch_bounds__(256) void router_k(const float* __restrict__ X,
                                                const float* __restrict__ Wr,
                                                float* __restrict__ logits) {
  __shared__ float sh[256];
  const int row = blockIdx.x, tid = threadIdx.x;
  const float* x = X + (size_t)row * H_;
  float s = 0.f;
  for (int i = tid; i < H_; i += 256) s += x[i] * Wr[i];
  sh[tid] = s; __syncthreads();
  for (int o = 128; o; o >>= 1) { if (tid < o) sh[tid] += sh[tid + o]; __syncthreads(); }
  if (tid == 0) logits[row] = sh[0];
}

// exact top-k by rank counting; tie-break by smaller index (matches lax.top_k)
__global__ __launch_bounds__(256) void mask_k(const float* __restrict__ logits,
                                              float* __restrict__ mask) {
  int idx = blockIdx.x * 256 + threadIdx.x;
  if (idx >= BS_) return;
  int b = idx / S_, s = idx % S_;
  const float* L = logits + (size_t)b * S_;
  float li = L[s];
  int rank = 0;
  for (int j = 0; j < S_; ++j) {
    float lj = L[j];
    rank += (lj > li) || ((lj == li) && (j < s));
  }
  mask[idx] = (rank < CAP_) ? 1.f : 0.f;
}

// ---------------- layernorm -> bf16 ----------------
__global__ __launch_bounds__(256) void layernorm_bf16_k(const float* __restrict__ X,
                                                        const float* __restrict__ g,
                                                        const float* __restrict__ bta,
                                                        __bf16* __restrict__ Y) {
  __shared__ float sh[256], sh2[256];
  const int row = blockIdx.x, tid = threadIdx.x;
  const float* x = X + (size_t)row * H_;
  float s = 0.f, s2 = 0.f;
  for (int i = tid; i < H_; i += 256) { float v = x[i]; s += v; s2 += v * v; }
  sh[tid] = s; sh2[tid] = s2; __syncthreads();
  for (int o = 128; o; o >>= 1) {
    if (tid < o) { sh[tid] += sh[tid + o]; sh2[tid] += sh2[tid + o]; }
    __syncthreads();
  }
  const float mu  = sh[0] * (1.f / H_);
  const float var = sh2[0] * (1.f / H_) - mu * mu;
  const float inv = rsqrtf(var + EPS_);
  for (int i = tid; i < H_; i += 256)
    Y[(size_t)row * H_ + i] = (__bf16)((x[i] - mu) * inv * g[i] + bta[i]);
}

// ---------------- WMMA GEMM: D[m,n] = sum_k A[m,k] * W[n,k] ----------------
// Block = 4 waves covering a 32x256 output tile. The 32x32 A tile per K-step is
// shared by all 4 waves: staged into LDS with async global->LDS copies (double
// buffered, ASYNCcnt + workgroup barrier), A fragments re-read via ds_load_b128.
// W fragments (per-wave unique) are direct global b128 loads.
// EP: 0 = store f32; 1 = store bf16; 2 = store bf16(silu(acc)); 3 = store bf16(acc * G[idx])
template <int EP>
__global__ __launch_bounds__(128) void gemm_bf16_k(const __bf16* __restrict__ A,
                                                   const __bf16* __restrict__ W,
                                                   void* __restrict__ Cout,
                                                   const __bf16* __restrict__ G,
                                                   int M, int N, int K) {
  __shared__ __align__(16) __bf16 atile[2][32 * 32];
  const int tid  = threadIdx.x;
  const int lane = tid & 31;
  const int wv   = tid >> 5;                    // wave in block: 0..3
  const int nb   = N >> 8;                      // 256-wide block tiles
  const int mbase = (int)(blockIdx.x / nb) << 5;
  const int nbase = ((int)(blockIdx.x % nb) << 8) + (wv << 6);

  const int lm  = lane & 15;
  const int klo = (lane < 16) ? 0 : 8;          // ISA 16-bit A/B fragment K-group layout
  const v8f z8  = {0.f, 0.f, 0.f, 0.f, 0.f, 0.f, 0.f, 0.f};

  // cooperative A staging: 128 threads x 16B = 32 rows x 32 cols bf16
  const int arow = tid >> 2;
  const int acol = (tid & 3) << 3;
  const __bf16* abase = A + (size_t)(mbase + arow) * K + acol;
  __bf16* lbase0 = &atile[0][arow * 32 + acol];
  __bf16* lbase1 = &atile[1][arow * 32 + acol];

  v8f acc[2][4];
#pragma unroll
  for (int i = 0; i < 2; ++i)
#pragma unroll
    for (int j = 0; j < 4; ++j) acc[i][j] = z8;

  stage16(abase, lbase0);                       // prologue: issue K-step 0
  int buf = 0;
  for (int kb = 0; kb < K; kb += 32) {
    stage_wait();
    __syncthreads();                            // current A tile resident for all waves
    if (kb + 32 < K) stage16(abase + kb + 32, buf ? lbase0 : lbase1);
    if (kb + 64 < K)
      __builtin_prefetch(W + (size_t)(nbase + lm) * K + kb + 64, 0, 0);

    FragBF a[2], w[4];
#pragma unroll
    for (int i = 0; i < 2; ++i) {               // A fragments from LDS
      const __bf16* p = &atile[buf][(16 * i + lm) * 32 + klo];
      a[i].q[0] = *(const uint4*)(p);           // K = klo .. klo+7
      a[i].q[1] = *(const uint4*)(p + 16);      // K = klo+16 .. klo+23
    }
#pragma unroll
    for (int j = 0; j < 4; ++j) {               // W fragments direct from global
      const __bf16* p = W + (size_t)(nbase + 16 * j + lm) * K + kb + klo;
      w[j].q[0] = *(const uint4*)(p);
      w[j].q[1] = *(const uint4*)(p + 16);
    }
#pragma unroll
    for (int i = 0; i < 2; ++i)
#pragma unroll
      for (int j = 0; j < 4; ++j)
        acc[i][j] = wmma_bf16f32(a[i].v, w[j].v, acc[i][j]);
    buf ^= 1;
  }

  const int halfo = (lane >> 4) << 3;           // upper half lanes hold rows +8
#pragma unroll
  for (int i = 0; i < 2; ++i) {
#pragma unroll
    for (int j = 0; j < 4; ++j) {
#pragma unroll
      for (int r = 0; r < 8; ++r) {
        const size_t row = (size_t)(mbase + 16 * i + r + halfo);
        const size_t col = (size_t)(nbase + 16 * j + lm);
        const size_t idx = row * (size_t)N + col;
        const float v = acc[i][j][r];
        if (EP == 0) {
          ((float*)Cout)[idx] = v;
        } else if (EP == 1) {
          ((__bf16*)Cout)[idx] = (__bf16)v;
        } else if (EP == 2) {
          ((__bf16*)Cout)[idx] = (__bf16)(v / (1.f + __expf(-v)));     // silu
        } else {
          ((__bf16*)Cout)[idx] = (__bf16)(v * (float)G[idx]);          // gate * up
        }
      }
    }
  }
}

// ---------------- flash attention (causal, 16 heads x 64) ----------------
// one wave per (batch, head, 16-query tile); bf16 WMMA, f32 softmax state
__global__ __launch_bounds__(128) void attn_k(const __bf16* __restrict__ Q,
                                              const __bf16* __restrict__ Kk,
                                              const __bf16* __restrict__ V,
                                              __bf16* __restrict__ O) {
  __shared__ __align__(16) __bf16 smem[4 * 16 * 32];  // per-wave 16x32 P tile
  const int lane = threadIdx.x & 31;
  const int wIn  = threadIdx.x >> 5;
  const int wid  = blockIdx.x * 4 + wIn;
  const int qtiles = S_ / 16;  // 128
  if (wid >= B_ * NH_ * qtiles) return;
  const int b  = wid / (NH_ * qtiles);
  const int rm = wid % (NH_ * qtiles);
  const int h  = rm / qtiles;
  const int qt = rm % qtiles;
  const int qbase = qt * 16;

  const size_t base = (size_t)b * S_ * H_ + (size_t)h * HD_;
  const int lm    = lane & 15;
  const int klo   = (lane < 16) ? 0 : 8;
  const int halfo = (lane >> 4) << 3;
  __bf16* pt = smem + wIn * (16 * 32);

  FragBF qf[2];
#pragma unroll
  for (int s = 0; s < 2; ++s) {
    const __bf16* p = Q + base + (size_t)(qbase + lm) * H_ + 32 * s + klo;
    qf[s].q[0] = *(const uint4*)p;
    qf[s].q[1] = *(const uint4*)(p + 16);
  }

  const v8f z8 = {0.f, 0.f, 0.f, 0.f, 0.f, 0.f, 0.f, 0.f};
  v8f oacc[4];
#pragma unroll
  for (int c = 0; c < 4; ++c) oacc[c] = z8;
  float rmax[8], rsum[8];
#pragma unroll
  for (int r = 0; r < 8; ++r) { rmax[r] = -__builtin_inff(); rsum[r] = 0.f; }

  const float sc = 0.125f;  // 1/sqrt(64)
  const int kend = qbase + 16;
  for (int t = 0; t < kend; t += 32) {
    // ---- S = Q K^T for 32 keys (2 chunks of 16), chained over d = 0..63 ----
    v8f sacc[2];
#pragma unroll
    for (int c = 0; c < 2; ++c) {
      sacc[c] = z8;
#pragma unroll
      for (int s = 0; s < 2; ++s) {
        FragBF kf;
        const __bf16* p = Kk + base + (size_t)(t + 16 * c + lm) * H_ + 32 * s + klo;
        kf.q[0] = *(const uint4*)p;
        kf.q[1] = *(const uint4*)(p + 16);
        sacc[c] = wmma_bf16f32(qf[s].v, kf.v, sacc[c]);
      }
    }
    // ---- causal mask + online softmax, stage P tile into LDS as bf16 ----
#pragma unroll
    for (int r = 0; r < 8; ++r) {
      const int qrow = qbase + r + halfo;
      float a0 = sacc[0][r] * sc;
      float a1 = sacc[1][r] * sc;
      if (t + lm      > qrow) a0 = -__builtin_inff();
      if (t + 16 + lm > qrow) a1 = -__builtin_inff();
      const float tmax = halfmax16(fmaxf(a0, a1));
      const float nm   = fmaxf(rmax[r], tmax);
      const float corr = __expf(rmax[r] - nm);
      rmax[r] = nm;
      const float p0 = __expf(a0 - nm);
      const float p1 = __expf(a1 - nm);
      rsum[r] = rsum[r] * corr + halfsum16(p0 + p1);
#pragma unroll
      for (int c4 = 0; c4 < 4; ++c4) oacc[c4][r] *= corr;
      pt[(r + halfo) * 32 + lm]      = (__bf16)p0;
      pt[(r + halfo) * 32 + 16 + lm] = (__bf16)p1;
    }
    asm volatile("s_wait_dscnt 0" ::: "memory");
    // ---- reload P in A-fragment layout; O += P V ----
    FragBF pf;
    pf.q[0] = *(const uint4*)(pt + lm * 32 + klo);
    pf.q[1] = *(const uint4*)(pt + lm * 32 + klo + 16);
#pragma unroll
    for (int c4 = 0; c4 < 4; ++c4) {
      FragBF vf;
#pragma unroll
      for (int j = 0; j < 8; ++j) {
        vf.v[j]     = V[base + (size_t)(t + klo + j) * H_      + c4 * 16 + lm];
        vf.v[j + 8] = V[base + (size_t)(t + klo + 16 + j) * H_ + c4 * 16 + lm];
      }
      oacc[c4] = wmma_bf16f32(pf.v, vf.v, oacc[c4]);
    }
  }
  // ---- normalize and write concat-head output as bf16 ----
#pragma unroll
  for (int c4 = 0; c4 < 4; ++c4)
#pragma unroll
    for (int r = 0; r < 8; ++r) {
      const float o = oacc[c4][r] / rsum[r];
      O[base + (size_t)(qbase + r + halfo) * H_ + c4 * 16 + lm] = (__bf16)o;
    }
}

// ---------------- host orchestration ----------------
extern "C" void kernel_launch(void* const* d_in, const int* in_sizes, int n_in,
                              void* d_out, int out_size, void* d_ws, size_t ws_size,
                              hipStream_t stream) {
  const float* x  = (const float*)d_in[0];
  const float* wr = (const float*)d_in[1];
  const float* g1 = (const float*)d_in[2];
  const float* b1 = (const float*)d_in[3];
  const float* wq = (const float*)d_in[4];
  const float* wk = (const float*)d_in[5];
  const float* wv = (const float*)d_in[6];
  const float* wo = (const float*)d_in[7];
  const float* g2 = (const float*)d_in[8];
  const float* b2 = (const float*)d_in[9];
  const float* wg = (const float*)d_in[10];
  const float* wu = (const float*)d_in[11];
  const float* wd = (const float*)d_in[12];

  float* out_x      = (float*)d_out;
  float* out_logits = out_x + BSH_;

  size_t off = 0;
  auto take = [&](size_t bytes) -> void* {
    void* p = (char*)d_ws + off;
    off += (bytes + 255) & ~(size_t)255;
    return p;
  };
  __bf16* wqb  = (__bf16*)take((size_t)H_ * H_ * 2);
  __bf16* wkb  = (__bf16*)take((size_t)H_ * H_ * 2);
  __bf16* wvb  = (__bf16*)take((size_t)H_ * H_ * 2);
  __bf16* wob  = (__bf16*)take((size_t)H_ * H_ * 2);
  __bf16* wgb  = (__bf16*)take((size_t)F_ * H_ * 2);
  __bf16* wub  = (__bf16*)take((size_t)F_ * H_ * 2);
  __bf16* wdb  = (__bf16*)take((size_t)H_ * F_ * 2);
  float*  mskp = (float*)take((size_t)BS_ * 4);
  __bf16* xnb  = (__bf16*)take(BSH_ * 2);
  __bf16* qb   = (__bf16*)take(BSH_ * 2);
  __bf16* kb   = (__bf16*)take(BSH_ * 2);
  __bf16* vb   = (__bf16*)take(BSH_ * 2);
  __bf16* aob  = (__bf16*)take(BSH_ * 2);
  float*  c32  = (float*)take(BSH_ * 4);   // WO out, then FFN-down out
  float*  x1   = (float*)take(BSH_ * 4);
  __bf16* gb   = (__bf16*)take(BSF_ * 2);  // silu(gate)
  __bf16* hb   = (__bf16*)take(BSF_ * 2);  // silu(gate)*up

  auto castN = [&](const float* src, __bf16* dst, size_t n) {
    cast_bf16_k<<<(unsigned)((n + 255) / 256), 256, 0, stream>>>(src, dst, n);
  };
  castN(wq, wqb, (size_t)H_ * H_);
  castN(wk, wkb, (size_t)H_ * H_);
  castN(wv, wvb, (size_t)H_ * H_);
  castN(wo, wob, (size_t)H_ * H_);
  castN(wg, wgb, (size_t)F_ * H_);
  castN(wu, wub, (size_t)F_ * H_);
  castN(wd, wdb, (size_t)H_ * F_);

  // router logits + top-k routing mask
  router_k<<<BS_, 256, 0, stream>>>(x, wr, out_logits);
  mask_k<<<BS_ / 256, 256, 0, stream>>>(out_logits, mskp);

  // grid: (M/32) x (N/256) blocks of 128 threads (4 waves x 64 cols)
  auto gemmGrid = [](int M, int N) { return dim3((unsigned)((M / 32) * (N / 256))); };

  // attention sub-block
  layernorm_bf16_k<<<BS_, 256, 0, stream>>>(x, g1, b1, xnb);
  gemm_bf16_k<1><<<gemmGrid(BS_, H_), 128, 0, stream>>>(xnb, wqb, qb, nullptr, BS_, H_, H_);
  gemm_bf16_k<1><<<gemmGrid(BS_, H_), 128, 0, stream>>>(xnb, wkb, kb, nullptr, BS_, H_, H_);
  gemm_bf16_k<1><<<gemmGrid(BS_, H_), 128, 0, stream>>>(xnb, wvb, vb, nullptr, BS_, H_, H_);
  attn_k<<<(B_ * NH_ * (S_ / 16)) / 4, 128, 0, stream>>>(qb, kb, vb, aob);
  gemm_bf16_k<0><<<gemmGrid(BS_, H_), 128, 0, stream>>>(aob, wob, c32, nullptr, BS_, H_, H_);
  resadd_k<<<(unsigned)(BSH_ / 256), 256, 0, stream>>>(x, c32, mskp, x1, BSH_);

  // FFN sub-block (SwiGLU)
  layernorm_bf16_k<<<BS_, 256, 0, stream>>>(x1, g2, b2, xnb);
  gemm_bf16_k<2><<<gemmGrid(BS_, F_), 128, 0, stream>>>(xnb, wgb, gb, nullptr, BS_, F_, H_);
  gemm_bf16_k<3><<<gemmGrid(BS_, F_), 128, 0, stream>>>(xnb, wub, hb, gb, BS_, F_, H_);
  gemm_bf16_k<0><<<gemmGrid(BS_, H_), 128, 0, stream>>>(hb, wdb, c32, nullptr, BS_, H_, F_);
  resadd_k<<<(unsigned)(BSH_ / 256), 256, 0, stream>>>(x1, c32, mskp, out_x, BSH_);
}
